// MultiHeadAttention_71528385348216
// MI455X (gfx1250) — matrix operand order repaired and checked by reading, not compiled, not run on hardware
//
#include <hip/hip_runtime.h>

typedef __attribute__((ext_vector_type(16))) _Float16 v16h;
typedef __attribute__((ext_vector_type(8)))  _Float16 v8h;
typedef __attribute__((ext_vector_type(8)))  float    v8f;

#define BATCH 4
#define SEQ   2048
#define DIM   768
#define HEADS 12
#define HD    64

// ---------------------------------------------------------------------------
// gfx1250 async global->LDS copy (ASYNCcnt-tracked, no VGPR data path).
// lds_off is a byte offset from the wave's LDS base (== dynamic-smem base
// when the kernel declares no static __shared__).
// ---------------------------------------------------------------------------
__device__ __forceinline__ void async_copy_b128(unsigned lds_off,
                                                const _Float16* gaddr) {
  asm volatile("global_load_async_to_lds_b128 %0, %1, off"
               :: "v"(lds_off), "v"(gaddr)
               : "memory");
}
__device__ __forceinline__ void wait_async0() {
  asm volatile("s_wait_asynccnt 0x0" ::: "memory");
}

// ---------------------------------------------------------------------------
// WMMA fragment loaders (wave32, CDNA5 ISA 7.12.2 layouts)
// ---------------------------------------------------------------------------
__device__ __forceinline__ v16h wmma_load_a(const _Float16* tile, int ld) {
  const int lane = threadIdx.x & 31;
  const int r    = lane & 15;
  const int kg   = (lane >> 4) << 3;       // 0 or 8
  const _Float16* base = tile + r * ld;
  v8h lo = *(const v8h*)(base + kg);       // K = kg .. kg+7
  v8h hi = *(const v8h*)(base + 16 + kg);  // K = 16+kg .. 16+kg+7
  v16h a;
#pragma unroll
  for (int i = 0; i < 8; ++i) { a[i] = lo[i]; a[8 + i] = hi[i]; }
  return a;
}

// B (32x16), tile stored N-major: tile[n*ld + k].
__device__ __forceinline__ v16h wmma_load_b(const _Float16* tile, int ld) {
  const int lane = threadIdx.x & 31;
  const int c    = lane & 15;
  const int kk   = (lane >> 4) << 3;       // 0 or 8
  const _Float16* base = tile + c * ld + kk;
  v8h lo = *(const v8h*)(base);
  v8h hi = *(const v8h*)(base + 16);
  v16h b;
#pragma unroll
  for (int i = 0; i < 8; ++i) { b[i] = lo[i]; b[8 + i] = hi[i]; }
  return b;
}

__device__ __forceinline__ v8f wmma16x16x32(v16h a, v16h b, v8f c) {
  return __builtin_amdgcn_wmma_f32_16x16x32_f16(false, a, false, b,
                                                (short)0, c, false, false);
}

// ---------------------------------------------------------------------------
// Kernel 1: QKV projection (f32 in -> f16 QKV ws). Staging needs f32->f16
// conversion, so it stays on the VALU path. Static LDS is fine here.
// ---------------------------------------------------------------------------
__global__ __launch_bounds__(256)
void qkv_kernel(const float* __restrict__ x, const float* __restrict__ wq,
                const float* __restrict__ bq, _Float16* __restrict__ Qo,
                _Float16* __restrict__ Ko, _Float16* __restrict__ Vo) {
  __shared__ __align__(16) _Float16 As[64 * 48];
  __shared__ __align__(16) _Float16 Bs[128 * 48];
  const int tid  = threadIdx.x;
  const int lane = tid & 31;
  const int wave = tid >> 5;
  const int bm = blockIdx.y * 64;
  const int bn = blockIdx.x * 128;
  const int wm = (wave >> 2) * 32;
  const int wn = (wave & 3) * 32;

  v8f acc[2][2] = {};

  for (int k0 = 0; k0 < DIM; k0 += 32) {
    __syncthreads();
#pragma unroll
    for (int i = 0; i < 2; ++i) {
      int idx = tid + i * 256;
      int row = idx >> 3, c4 = (idx & 7) * 4;
      const float4 f = *(const float4*)(x + (size_t)(bm + row) * DIM + k0 + c4);
      As[row * 48 + c4 + 0] = (_Float16)f.x;
      As[row * 48 + c4 + 1] = (_Float16)f.y;
      As[row * 48 + c4 + 2] = (_Float16)f.z;
      As[row * 48 + c4 + 3] = (_Float16)f.w;
    }
#pragma unroll
    for (int i = 0; i < 4; ++i) {
      int idx = tid + i * 256;
      int row = idx >> 5, c4 = (idx & 31) * 4;
      const float4 f =
          *(const float4*)(wq + (size_t)(k0 + row) * (3 * DIM) + bn + c4);
      Bs[(c4 + 0) * 48 + row] = (_Float16)f.x;
      Bs[(c4 + 1) * 48 + row] = (_Float16)f.y;
      Bs[(c4 + 2) * 48 + row] = (_Float16)f.z;
      Bs[(c4 + 3) * 48 + row] = (_Float16)f.w;
    }
    __syncthreads();
    v16h a0 = wmma_load_a(As + (wm + 0) * 48, 48);
    v16h a1 = wmma_load_a(As + (wm + 16) * 48, 48);
    v16h b0 = wmma_load_b(Bs + (wn + 0) * 48, 48);
    v16h b1 = wmma_load_b(Bs + (wn + 16) * 48, 48);
    acc[0][0] = wmma16x16x32(a0, b0, acc[0][0]);
    acc[0][1] = wmma16x16x32(a0, b1, acc[0][1]);
    acc[1][0] = wmma16x16x32(a1, b0, acc[1][0]);
    acc[1][1] = wmma16x16x32(a1, b1, acc[1][1]);
  }

  const float qscale = 0.28867513459481287f;  // 1/sqrt(12)
#pragma unroll
  for (int mi = 0; mi < 2; ++mi)
#pragma unroll
    for (int ni = 0; ni < 2; ++ni) {
      const int gcol = bn + wn + ni * 16 + (lane & 15);
      const float bias = bq[gcol];
      const int which = gcol / DIM;
      const int rem = gcol - which * DIM;
      const int h = rem >> 6, d = rem & 63;
      _Float16* dst = (which == 0) ? Qo : (which == 1) ? Ko : Vo;
      const float sc = (which == 0) ? qscale : 1.0f;
#pragma unroll
      for (int r = 0; r < 8; ++r) {
        const int grow = bm + wm + mi * 16 + r + ((lane >> 4) << 3);
        const int b = grow >> 11, s = grow & (SEQ - 1);
        const float v = (acc[mi][ni][r] + bias) * sc;
        dst[((size_t)(b * HEADS + h) * SEQ + s) * HD + d] = (_Float16)v;
      }
    }
}

// ---------------------------------------------------------------------------
// Kernel 2: flash attention, double-buffered key loop.
// Dynamic LDS layout (byte offsets, base 0 == dynamic smem base):
//   Ks[2]  : 2 * 4096  = [0, 8192)    32 keys x 64 dims, row-major (async)
//   Vt[2]  : 2 * 5120  = [8192,18432) 64 dims x (32+8) keys, transposed
//   Ps[4]  : 4 * 1280  = [18432,23552) per-wave P tiles, ld 40
// Pattern per iter: async-stage tile i+1 -> compute tile i -> wait -> barrier.
// ---------------------------------------------------------------------------
#define ATTN_SMEM_BYTES (2 * 4096 + 2 * 5120 + 4 * 1280)

__global__ __launch_bounds__(128)
void attn_kernel(const _Float16* __restrict__ Q, const _Float16* __restrict__ K,
                 const _Float16* __restrict__ V, _Float16* __restrict__ O) {
  extern __shared__ __align__(16) char smem[];
  _Float16* KsBuf = (_Float16*)smem;                        // 2 x 2048 halfs
  _Float16* VtBuf = (_Float16*)(smem + 2 * 4096);           // 2 x 2560 halfs
  _Float16* PsAll = (_Float16*)(smem + 2 * 4096 + 2 * 5120);

  const int tid  = threadIdx.x;
  const int lane = tid & 31;
  const int wave = tid >> 5;
  const int bh = blockIdx.y;
  const int b = bh / HEADS, h = bh - b * HEADS;
  const int q0 = blockIdx.x * 64 + wave * 16;

  const _Float16* Qb = Q + (size_t)bh * SEQ * HD;
  const _Float16* Kb = K + (size_t)bh * SEQ * HD;
  const _Float16* Vb = V + (size_t)bh * SEQ * HD;
  _Float16* Ps = PsAll + wave * 640;

  // Q fragments (16 rows x 64 dims), A layout straight from global
  v16h aq[2];
  {
    const int r = lane & 15, kg = (lane >> 4) << 3;
#pragma unroll
    for (int t = 0; t < 2; ++t) {
      const _Float16* base = Qb + (size_t)(q0 + r) * HD + t * 32;
      v8h lo = *(const v8h*)(base + kg);
      v8h hi = *(const v8h*)(base + 16 + kg);
#pragma unroll
      for (int i = 0; i < 8; ++i) { aq[t][i] = lo[i]; aq[t][8 + i] = hi[i]; }
    }
  }

  v8f oacc[4] = {};
  float m_r[8], l_r[8];
#pragma unroll
  for (int r = 0; r < 8; ++r) { m_r[r] = -1e30f; l_r[r] = 0.0f; }

  // stage key/value tiles for rows [k0, k0+32) into buffer `buf`
  auto stage_tiles = [&](int k0, int buf) {
    // K tile: f16->f16, linear -> async DMA, 2 x 16B per thread
#pragma unroll
    for (int i = 0; i < 2; ++i) {
      const int idx = tid + i * 128;
      const unsigned lds_off = (unsigned)(buf * 4096 + idx * 16);
      async_copy_b128(lds_off, Kb + (size_t)k0 * HD + idx * 8);
    }
    // V tile: needs transpose -> manual scatter
    _Float16* Vt = VtBuf + buf * 2560;
#pragma unroll
    for (int i = 0; i < 2; ++i) {
      const int idx = tid + i * 128;
      const int key = idx >> 3, d8 = (idx & 7) * 8;
      v8h v = *(const v8h*)(Vb + (size_t)(k0 + key) * HD + d8);
#pragma unroll
      for (int j = 0; j < 8; ++j) Vt[(d8 + j) * 40 + key] = v[j];
    }
  };

  stage_tiles(0, 0);
  wait_async0();
  __syncthreads();

  const int NIT = SEQ / 32;
  for (int it = 0; it < NIT; ++it) {
    const int buf = it & 1;
    const _Float16* Ks = KsBuf + buf * 2048;
    const _Float16* Vt = VtBuf + buf * 2560;

    // prefetch next tile into the other buffer (read at it-1, fenced by the
    // end-of-(it-1) barrier; every wave's frag loads were dscnt-waited
    // before its WMMAs, which precede that barrier)
    if (it + 1 < NIT) stage_tiles((it + 1) * 32, buf ^ 1);

    // scores: 16 q-rows x 32 keys (Q pre-scaled by 1/sqrt(H))
    v8f c0 = {}, c1 = {};
#pragma unroll
    for (int t = 0; t < 2; ++t) {
      v16h bk0 = wmma_load_b(Ks + 0 * 64 + t * 32, 64);   // keys 0..15
      v16h bk1 = wmma_load_b(Ks + 16 * 64 + t * 32, 64);  // keys 16..31
      c0 = wmma16x16x32(aq[t], bk0, c0);
      c1 = wmma16x16x32(aq[t], bk1, c1);
    }

    // online softmax; row reductions across 16-lane halves
    const int half = (lane >> 4) << 3;
#pragma unroll
    for (int r = 0; r < 8; ++r) {
      float s0 = c0[r], s1 = c1[r];
      float mx = fmaxf(s0, s1);
      mx = fmaxf(mx, __shfl_xor(mx, 1));
      mx = fmaxf(mx, __shfl_xor(mx, 2));
      mx = fmaxf(mx, __shfl_xor(mx, 4));
      mx = fmaxf(mx, __shfl_xor(mx, 8));
      const float mn = fmaxf(m_r[r], mx);
      const float p0 = __expf(s0 - mn);
      const float p1 = __expf(s1 - mn);
      float rs = p0 + p1;
      rs += __shfl_xor(rs, 1);
      rs += __shfl_xor(rs, 2);
      rs += __shfl_xor(rs, 4);
      rs += __shfl_xor(rs, 8);
      const float corr = __expf(m_r[r] - mn);
      l_r[r] = l_r[r] * corr + rs;
      m_r[r] = mn;
#pragma unroll
      for (int t = 0; t < 4; ++t) oacc[t][r] *= corr;
      const int row = r + half;
      Ps[row * 40 + (lane & 15)]      = (_Float16)p0;
      Ps[row * 40 + 16 + (lane & 15)] = (_Float16)p1;
    }
    // cross-lane P store -> A-layout reload (wave-ordered DS; force wait)
    asm volatile("s_wait_dscnt 0x0" ::: "memory");
    v16h ap = wmma_load_a(Ps, 40);
#pragma unroll
    for (int t = 0; t < 4; ++t) {
      v16h bv = wmma_load_b(Vt + (t * 16) * 40, 40);
      oacc[t] = wmma16x16x32(ap, bv, oacc[t]);
    }

    wait_async0();      // prefetch landed in LDS
    __syncthreads();    // visible to all waves
  }

  // normalize and store merged-head f16 [B, S, H*64]
  const int half = (lane >> 4) << 3;
#pragma unroll
  for (int r = 0; r < 8; ++r) {
    const float inv = 1.0f / l_r[r];
    const int row = q0 + r + half;
    const size_t base = ((size_t)(b * SEQ + row)) * DIM + h * HD;
#pragma unroll
    for (int t = 0; t < 4; ++t)
      O[base + t * 16 + (lane & 15)] = (_Float16)(oacc[t][r] * inv);
  }
}

// ---------------------------------------------------------------------------
// Kernel 3: output projection. A tile is already f16 -> async DMA staging.
// Dynamic LDS: As [0, 6144), Bs [6144, 18432).
// ---------------------------------------------------------------------------
#define PROJ_SMEM_BYTES (64 * 48 * 2 + 128 * 48 * 2)

__global__ __launch_bounds__(256)
void out_proj_kernel(const _Float16* __restrict__ A, const float* __restrict__ w,
                     const float* __restrict__ bias, float* __restrict__ out) {
  extern __shared__ __align__(16) char smem[];
  _Float16* As = (_Float16*)smem;             // 64 x 48
  _Float16* Bs = (_Float16*)(smem + 6144);    // 128 x 48
  const int tid  = threadIdx.x;
  const int lane = tid & 31;
  const int wave = tid >> 5;
  const int bm = blockIdx.y * 64;
  const int bn = blockIdx.x * 128;
  const int wm = (wave >> 2) * 32;
  const int wn = (wave & 3) * 32;

  v8f acc[2][2] = {};

  for (int k0 = 0; k0 < DIM; k0 += 32) {
    __syncthreads();
    {  // A tile: f16->f16, one async 16B chunk per thread
      const int row = tid >> 2, c8 = (tid & 3) * 8;
      const unsigned lds_off = (unsigned)((row * 48 + c8) * 2);
      async_copy_b128(lds_off, A + (size_t)(bm + row) * DIM + k0 + c8);
    }
#pragma unroll
    for (int i = 0; i < 4; ++i) {  // w tile: f32->f16 transposed (VALU path)
      int idx = tid + i * 256;
      int row = idx >> 5, c4 = (idx & 31) * 4;
      const float4 f = *(const float4*)(w + (size_t)(k0 + row) * DIM + bn + c4);
      Bs[(c4 + 0) * 48 + row] = (_Float16)f.x;
      Bs[(c4 + 1) * 48 + row] = (_Float16)f.y;
      Bs[(c4 + 2) * 48 + row] = (_Float16)f.z;
      Bs[(c4 + 3) * 48 + row] = (_Float16)f.w;
    }
    wait_async0();
    __syncthreads();
    v16h a0 = wmma_load_a(As + (wm + 0) * 48, 48);
    v16h a1 = wmma_load_a(As + (wm + 16) * 48, 48);
    v16h b0 = wmma_load_b(Bs + (wn + 0) * 48, 48);
    v16h b1 = wmma_load_b(Bs + (wn + 16) * 48, 48);
    acc[0][0] = wmma16x16x32(a0, b0, acc[0][0]);
    acc[0][1] = wmma16x16x32(a0, b1, acc[0][1]);
    acc[1][0] = wmma16x16x32(a1, b0, acc[1][0]);
    acc[1][1] = wmma16x16x32(a1, b1, acc[1][1]);
  }

#pragma unroll
  for (int mi = 0; mi < 2; ++mi)
#pragma unroll
    for (int ni = 0; ni < 2; ++ni) {
      const int gcol = bn + wn + ni * 16 + (lane & 15);
      const float bv = bias[gcol];
#pragma unroll
      for (int r = 0; r < 8; ++r) {
        const int grow = bm + wm + mi * 16 + r + ((lane >> 4) << 3);
        out[(size_t)grow * DIM + gcol] = acc[mi][ni][r] + bv;
      }
    }
}

// ---------------------------------------------------------------------------
extern "C" void kernel_launch(void* const* d_in, const int* in_sizes, int n_in,
                              void* d_out, int out_size, void* d_ws,
                              size_t ws_size, hipStream_t stream) {
  const float* x     = (const float*)d_in[0];
  const float* w_qkv = (const float*)d_in[1];
  const float* b_qkv = (const float*)d_in[2];
  const float* w_out = (const float*)d_in[3];
  const float* b_out = (const float*)d_in[4];
  float* out = (float*)d_out;

  // workspace: Q,K,V [B,H,S,64] f16 + attn_out [B,S,768] f16 (~50.3 MB)
  const size_t perbuf = (size_t)BATCH * HEADS * SEQ * HD;  // 6,291,456 elems
  _Float16* Qw = (_Float16*)d_ws;
  _Float16* Kw = Qw + perbuf;
  _Float16* Vw = Kw + perbuf;
  _Float16* Aw = Vw + perbuf;

  qkv_kernel<<<dim3((3 * DIM) / 128, (BATCH * SEQ) / 64), 256, 0, stream>>>(
      x, w_qkv, b_qkv, Qw, Kw, Vw);
  attn_kernel<<<dim3(SEQ / 64, BATCH * HEADS), 128, ATTN_SMEM_BYTES, stream>>>(
      Qw, Kw, Vw, Aw);
  out_proj_kernel<<<dim3(DIM / 128, (BATCH * SEQ) / 64), 256, PROJ_SMEM_BYTES,
                    stream>>>(Aw, w_out, b_out, out);
}
